// QuadraticAttention_53815940219149
// MI455X (gfx1250) — matrix operand order repaired and checked by reading, not compiled, not run on hardware
//
#include <hip/hip_runtime.h>
#include <cstdint>
#include <cstddef>

#define D_MODEL 1024
#define N_HEAD  16
#define D_HEAD  64
#define N_CTX   2048
#define BATCH   2
#define M_ROWS  (BATCH * N_CTX)   // 4096

typedef __attribute__((ext_vector_type(16))) _Float16 v16h;
typedef __attribute__((ext_vector_type(8)))  float    v8f;

union HFrag { v16h h; uint4 q[2]; };

static __device__ __forceinline__ v8f wmma16(v16h a, v16h b, v8f c) {
    // v_wmma_f32_16x16x32_f16 : (neg_a, A, neg_b, B, c_mod, C, reuse_a, reuse_b)
    return __builtin_amdgcn_wmma_f32_16x16x32_f16(false, a, false, b, (short)0, c, false, false);
}

// A-matrix (16x32 f16, M x K): lane = m + 16*g; VGPR v holds K = (v&3)*2 + g*8 + (v>>2)*16 (+h)
static __device__ __forceinline__ v16h load_fragA(const _Float16* p, int stride_h, int lane) {
    const int m = lane & 15, g = lane >> 4;
    HFrag f;
    const char* b = (const char*)(p + (size_t)m * stride_h + g * 8);
    f.q[0] = *(const uint4*)(b);        // K = g*8 + 0..7
    f.q[1] = *(const uint4*)(b + 32);   // K = g*8 + 16..23
    return f.h;
}

// B-matrix (32x16 f16, K x N): lane = n + 16*g; VGPR v holds K = (v&3)*2 + g*8 + (v>>2)*16 (+h)
static __device__ __forceinline__ v16h load_fragB(const _Float16* p, int stride_h, int lane) {
    const int n = lane & 15, g = lane >> 4;
    HFrag f;
    const char* b = (const char*)(p + (size_t)n * stride_h + g * 8);
    f.q[0] = *(const uint4*)(b);        // K = g*8 + 0..7
    f.q[1] = *(const uint4*)(b + 32);   // K = g*8 + 16..23
    return f.h;
}

// 16B global -> LDS async copy (ASYNCcnt-tracked, no VGPR round-trip).
// Generic pointer to __shared__ keeps the wave-relative LDS byte offset in addr[31:0].
static __device__ __forceinline__ void async_copy16(const _Float16* g, _Float16* l) {
    const uint32_t lds_off = (uint32_t)(uintptr_t)l;
    asm volatile("global_load_async_to_lds_b128 %0, %1, off"
                 :: "v"(lds_off), "v"(g) : "memory");
}

// ---------------------------------------------------------------------------
// f32 -> f16 bulk convert (x and the four weight matrices, done once)
// ---------------------------------------------------------------------------
__global__ __launch_bounds__(256)
void cvt_f16_kernel(const float* __restrict__ src, _Float16* __restrict__ dst, int n4) {
    const int i = blockIdx.x * blockDim.x + threadIdx.x;
    if (i < n4) {
        const float4 v = ((const float4*)src)[i];
        _Float16* d = dst + (size_t)i * 4;
        d[0] = (_Float16)v.x; d[1] = (_Float16)v.y;
        d[2] = (_Float16)v.z; d[3] = (_Float16)v.w;
    }
}

// ---------------------------------------------------------------------------
// GEMM: out[M,1024] = Ah[M,K] @ Wh^T (+bias) (+residual), Ah/Wh f16.
// 64x64 tile / 128-thread block, async double-buffered LDS.
// ---------------------------------------------------------------------------
#define TILE_M 64
#define TILE_N 64
#define TILE_K 32
#define LDSS   40   // padded f16 stride: 80B rows keep 16B alignment, dodge bank conflicts

template <bool HAS_BIAS, bool HAS_RES>
__global__ __launch_bounds__(128)
void gemm_f16_kernel(const _Float16* __restrict__ Ah, const _Float16* __restrict__ Wh,
                     const float* __restrict__ bias, const float* __restrict__ residual,
                     float* __restrict__ out, int K) {
    __shared__ __align__(16) _Float16 As[2][TILE_M * LDSS];
    __shared__ __align__(16) _Float16 Bs[2][TILE_N * LDSS];

    const int tid  = threadIdx.x;
    const int lane = tid & 31;
    const int wave = tid >> 5;
    const int row0 = blockIdx.y * TILE_M;
    const int col0 = blockIdx.x * TILE_N;

    // Per tile: 64 rows x 64B = 256 16B-chunks per matrix; 2 A + 2 B chunks/thread.
    const int r0 = (tid + 0)   >> 2, c0 = ((tid + 0)   & 3) * 8;
    const int r1 = (tid + 128) >> 2, c1 = ((tid + 128) & 3) * 8;

    auto issue_tile = [&](int k0, int buf) {
        async_copy16(Ah + (size_t)(row0 + r0) * K + k0 + c0, &As[buf][r0 * LDSS + c0]);
        async_copy16(Ah + (size_t)(row0 + r1) * K + k0 + c1, &As[buf][r1 * LDSS + c1]);
        async_copy16(Wh + (size_t)(col0 + r0) * K + k0 + c0, &Bs[buf][r0 * LDSS + c0]);
        async_copy16(Wh + (size_t)(col0 + r1) * K + k0 + c1, &Bs[buf][r1 * LDSS + c1]);
    };

    v8f acc[4] = {};
    const int nk = K / TILE_K;
    issue_tile(0, 0);
    for (int kt = 0; kt < nk; ++kt) {
        const int cb = kt & 1;
        if (kt + 1 < nk) {
            issue_tile((kt + 1) * TILE_K, cb ^ 1);
            // 4 in flight for the next tile; retire this tile's 4 (in-order completion)
            asm volatile("s_wait_asynccnt 4" ::: "memory");
        } else {
            asm volatile("s_wait_asynccnt 0" ::: "memory");
        }
        __syncthreads();
        // gather all fragments first -> single dscnt wait, then 4 back-to-back WMMAs
        const v16h af = load_fragA(&As[cb][(wave * 16) * LDSS], LDSS, lane);
        v16h bf[4];
        #pragma unroll
        for (int nt = 0; nt < 4; ++nt)
            bf[nt] = load_fragB(&Bs[cb][(nt * 16) * LDSS], LDSS, lane);
        #pragma unroll
        for (int nt = 0; nt < 4; ++nt)
            acc[nt] = wmma16(af, bf[nt], acc[nt]);
        __syncthreads();   // all waves done reading before buffer is overwritten
    }

    // C layout: lane = n + 16*g ; VGPR r -> row m = r + 8*g
    const int n_lo = lane & 15, g = lane >> 4;
    #pragma unroll
    for (int nt = 0; nt < 4; ++nt) {
        const int gcol = col0 + nt * 16 + n_lo;
        const float badd = HAS_BIAS ? bias[gcol] : 0.0f;
        #pragma unroll
        for (int r = 0; r < 8; ++r) {
            const int grow = row0 + wave * 16 + r + 8 * g;
            float v = acc[nt][r] + badd;
            if (HAS_RES) v += residual[(size_t)grow * D_MODEL + gcol];
            out[(size_t)grow * D_MODEL + gcol] = v;
        }
    }
}

// ---------------------------------------------------------------------------
// RMSNorm + RoPE on q,k ; pack q,k -> f16 [b,h,s,d] ; v -> f16 transposed [b,h,d,s]
// ---------------------------------------------------------------------------
__global__ __launch_bounds__(256)
void normrope_pack_kernel(const float* __restrict__ q32, const float* __restrict__ k32,
                          const float* __restrict__ v32, const float* __restrict__ nw,
                          _Float16* __restrict__ qh, _Float16* __restrict__ kh,
                          _Float16* __restrict__ vT) {
    const int t = blockIdx.x * blockDim.x + threadIdx.x;  // 0..65535 = (b,s,h)
    const int h  = t & 15;
    const int sb = t >> 4;
    const int s  = sb & (N_CTX - 1);
    const int b  = sb >> 11;
    const int bh = b * N_HEAD + h;

    const size_t src = ((size_t)(b * N_CTX + s)) * D_MODEL + (size_t)h * D_HEAD;
    const size_t dst = ((size_t)bh * N_CTX + s) * D_HEAD;
    const size_t vtb = ((size_t)bh * D_HEAD) * N_CTX;

    float sq = 0.0f, sk = 0.0f;
    for (int d = 0; d < D_HEAD; ++d) {
        const float a = q32[src + d], c = k32[src + d];
        sq += a * a; sk += c * c;
    }
    const float eps = 1.1920929e-07f;
    const float rq = rsqrtf(sq * (1.0f / D_HEAD) + eps);
    const float rk = rsqrtf(sk * (1.0f / D_HEAD) + eps);

    const float pos = (float)s;
    const float LN10K = 9.210340371976184f;  // ln(10000)
    for (int i = 0; i < 32; ++i) {
        const int d0 = 2 * i, d1 = 2 * i + 1;
        const int j0 = d0 & 31, j1 = d1 & 31;
        const float a0 = pos * __expf(-(2.0f * j0 / (float)D_HEAD) * LN10K);
        const float a1 = pos * __expf(-(2.0f * j1 / (float)D_HEAD) * LN10K);
        const float c0 = __cosf(a0), s0 = __sinf(a0);
        const float c1 = __cosf(a1), s1 = __sinf(a1);
        const float w0 = nw[d0], w1 = nw[d1];

        const float nq0 = q32[src + d0] * rq * w0, nq1 = q32[src + d1] * rq * w1;
        const float nk0 = k32[src + d0] * rk * w0, nk1 = k32[src + d1] * rk * w1;

        qh[dst + d0] = (_Float16)(nq0 * c0 - nq1 * s0);
        qh[dst + d1] = (_Float16)(nq1 * c1 + nq0 * s1);
        kh[dst + d0] = (_Float16)(nk0 * c0 - nk1 * s0);
        kh[dst + d1] = (_Float16)(nk1 * c1 + nk0 * s1);
    }
    for (int d = 0; d < D_HEAD; ++d)
        vT[vtb + (size_t)d * N_CTX + s] = (_Float16)v32[src + d];
}

// ---------------------------------------------------------------------------
// Attention: per (b,h, 16-query tile) wave. scores=q.k ; p=(s/64)^2*causal ; z=p@v
// Emits z directly in f16 so the Wo projection runs the all-f16 async GEMM.
// ---------------------------------------------------------------------------
__global__ __launch_bounds__(32)
void attn_kernel(const _Float16* __restrict__ qh, const _Float16* __restrict__ kh,
                 const _Float16* __restrict__ vT, _Float16* __restrict__ zh) {
    __shared__ __align__(16) _Float16 Ps[16 * LDSS];   // 16 x 32 pattern tile

    const int lane = threadIdx.x;
    const int qt = blockIdx.x;
    const int bh = blockIdx.y;
    const int q0 = qt * 16;
    const int m  = lane & 15, g = lane >> 4;

    const size_t seq_base = (size_t)bh * N_CTX * D_HEAD;   // qh/kh [b,h,s,d]
    const size_t vt_base  = (size_t)bh * D_HEAD * N_CTX;   // vT    [b,h,d,s]

    const v16h qa0 = load_fragA(qh + seq_base + (size_t)q0 * D_HEAD + 0,  D_HEAD, lane);
    const v16h qa1 = load_fragA(qh + seq_base + (size_t)q0 * D_HEAD + 32, D_HEAD, lane);

    v8f z[4] = {};
    const int npair = (qt >> 1) + 1;        // 32-key blocks covering causal range
    for (int kp = 0; kp < npair; ++kp) {
        const int kbase = kp * 32;
        #pragma unroll
        for (int half = 0; half < 2; ++half) {
            const int k0 = kbase + half * 16;
            const _Float16* kt = kh + seq_base + (size_t)k0 * D_HEAD;
            // both K-fragments up front, then the WMMA pair
            const v16h kb0 = load_fragB(kt,      D_HEAD, lane);
            const v16h kb1 = load_fragB(kt + 32, D_HEAD, lane);
            v8f s = {};
            s = wmma16(qa0, kb0, s);
            s = wmma16(qa1, kb1, s);
            const int keyg = k0 + m;        // score C layout: col = key
            #pragma unroll
            for (int r = 0; r < 8; ++r) {
                const int qg = q0 + r + 8 * g;
                float sv = s[r] * (1.0f / (float)D_HEAD);
                sv = sv * sv;
                if (keyg > qg) sv = 0.0f;
                Ps[(r + 8 * g) * LDSS + half * 16 + m] = (_Float16)sv;
            }
        }
        __syncthreads();
        const v16h pa = load_fragA(Ps, LDSS, lane);      // K = 32 keys
        v16h vb[4];
        #pragma unroll
        for (int nt = 0; nt < 4; ++nt)
            vb[nt] = load_fragB(vT + vt_base + (size_t)(nt * 16) * N_CTX + kbase, N_CTX, lane);
        #pragma unroll
        for (int nt = 0; nt < 4; ++nt)
            z[nt] = wmma16(pa, vb[nt], z[nt]);
        __syncthreads();
    }

    const int b = bh >> 4, h = bh & 15;
    #pragma unroll
    for (int nt = 0; nt < 4; ++nt) {
        const int dcol = h * D_HEAD + nt * 16 + m;
        #pragma unroll
        for (int r = 0; r < 8; ++r) {
            const int srow = q0 + r + 8 * g;
            zh[(size_t)(b * N_CTX + srow) * D_MODEL + dcol] = (_Float16)z[nt][r];
        }
    }
}

// ---------------------------------------------------------------------------
extern "C" void kernel_launch(void* const* d_in, const int* in_sizes, int n_in,
                              void* d_out, int out_size, void* d_ws, size_t ws_size,
                              hipStream_t stream) {
    (void)in_sizes; (void)n_in; (void)out_size; (void)ws_size;
    const float* x  = (const float*)d_in[0];
    const float* Wq = (const float*)d_in[1];
    const float* bq = (const float*)d_in[2];
    const float* Wk = (const float*)d_in[3];
    const float* bk = (const float*)d_in[4];
    const float* Wv = (const float*)d_in[5];
    const float* bv = (const float*)d_in[6];
    const float* Wo = (const float*)d_in[7];
    const float* nw = (const float*)d_in[8];
    float* out = (float*)d_out;

    const size_t E  = (size_t)M_ROWS * D_MODEL;   // 4,194,304
    const size_t WE = (size_t)D_MODEL * D_MODEL;  // 1,048,576

    _Float16* xh  = (_Float16*)d_ws;              //  8 MB
    _Float16* Wqh = xh  + E;                      //  2 MB each
    _Float16* Wkh = Wqh + WE;
    _Float16* Wvh = Wkh + WE;
    _Float16* Woh = Wvh + WE;
    float* q32 = (float*)(Woh + WE);              // 16 MB each
    float* k32 = q32 + E;
    float* v32 = k32 + E;
    _Float16* qh = (_Float16*)(v32 + E);          //  8 MB each
    _Float16* kh = qh + E;
    _Float16* vT = kh + E;
    _Float16* zh = (_Float16*)q32;                // reuse q32 after packing (total 88 MB)

    // one-shot f32 -> f16 of x and weights (weights reused 64x per GEMM)
    cvt_f16_kernel<<<dim3((int)(E  / 4 / 256)), 256, 0, stream>>>(x,  xh,  (int)(E  / 4));
    cvt_f16_kernel<<<dim3((int)(WE / 4 / 256)), 256, 0, stream>>>(Wq, Wqh, (int)(WE / 4));
    cvt_f16_kernel<<<dim3((int)(WE / 4 / 256)), 256, 0, stream>>>(Wk, Wkh, (int)(WE / 4));
    cvt_f16_kernel<<<dim3((int)(WE / 4 / 256)), 256, 0, stream>>>(Wv, Wvh, (int)(WE / 4));
    cvt_f16_kernel<<<dim3((int)(WE / 4 / 256)), 256, 0, stream>>>(Wo, Woh, (int)(WE / 4));

    const dim3 gg(D_MODEL / TILE_N, M_ROWS / TILE_M);   // (16, 64)
    gemm_f16_kernel<true,  false><<<gg, 128, 0, stream>>>(xh, Wqh, bq, nullptr, q32, D_MODEL);
    gemm_f16_kernel<true,  false><<<gg, 128, 0, stream>>>(xh, Wkh, bk, nullptr, k32, D_MODEL);
    gemm_f16_kernel<true,  false><<<gg, 128, 0, stream>>>(xh, Wvh, bv, nullptr, v32, D_MODEL);

    normrope_pack_kernel<<<dim3((BATCH * N_CTX * N_HEAD) / 256), 256, 0, stream>>>(
        q32, k32, v32, nw, qh, kh, vT);

    attn_kernel<<<dim3(N_CTX / 16, BATCH * N_HEAD), 32, 0, stream>>>(qh, kh, vT, zh);

    gemm_f16_kernel<false, true><<<gg, 128, 0, stream>>>(zh, Woh, nullptr, x, out, D_MODEL);
}
